// INT8Layer_66821101191782
// MI455X (gfx1250) — matrix operand order repaired and checked by reading, not compiled, not run on hardware
//
#include <hip/hip_runtime.h>
#include <stddef.h>

// ---------------------------------------------------------------------------
// int8-weight dense layer:  out = (x_f32 @ W_i8^T) * scales + bias
//   M = B*S = 8192, K = 4096, N = 4096  (AI ~965 FLOP/B -> WMMA-bound)
//
// Preferred plan (needs 96 MiB workspace):
//   pass 1a: x fp32 -> f16  (64 MiB, bandwidth-bound, ~3.5 us)
//   pass 1b: W int8 -> f16  (32 MiB, EXACT conversion, ~2 us)
//   pass 2 : pure-f16 double-buffered LDS GEMM, v_wmma_f32_16x16x32_f16,
//            BK=64 slab = 16 WMMAs per workgroup barrier, fp32 accum,
//            per-channel scale + bias epilogue.
// Fallback (small ws): fused conversion-in-loop GEMM (round-2 kernel).
// ---------------------------------------------------------------------------

typedef __attribute__((ext_vector_type(16))) _Float16 v16h;
typedef __attribute__((ext_vector_type(8)))  _Float16 v8h;
typedef __attribute__((ext_vector_type(4)))  _Float16 v4h;
typedef __attribute__((ext_vector_type(8)))  float    v8f;

#define DIN      4096
#define DOUT     4096
#define MTOTAL   8192
#define BM       128
#define BN       128
#define BK       64             // staged K-slab (2 x WMMA K of 32)
#define NK       (DIN / BK)     // 64 staged iterations
#define LDS_PAD  8
#define LDS_LD   (BK + LDS_PAD) // 72 halves per row (144 B, 16B-aligned)

union FragU { v16h v; v8h h[2]; };

// ---------------------------------------------------------------------------
// Pass 1a: fp32 -> f16 (8 elements / thread, b128 in, b128 out)
// ---------------------------------------------------------------------------
__global__ __launch_bounds__(256)
void cvt_x_f16(const float* __restrict__ x, _Float16* __restrict__ xa)
{
    const size_t i0 = ((size_t)blockIdx.x * 256 + threadIdx.x) * 8;
    float4 f0 = *(const float4*)(x + i0);
    float4 f1 = *(const float4*)(x + i0 + 4);
    v8h h;
    h[0] = (_Float16)f0.x; h[1] = (_Float16)f0.y;
    h[2] = (_Float16)f0.z; h[3] = (_Float16)f0.w;
    h[4] = (_Float16)f1.x; h[5] = (_Float16)f1.y;
    h[6] = (_Float16)f1.z; h[7] = (_Float16)f1.w;
    *(v8h*)(xa + i0) = h;
}

// ---------------------------------------------------------------------------
// Pass 1b: int8 -> f16 exact (16 elements / thread)
// ---------------------------------------------------------------------------
__global__ __launch_bounds__(256)
void cvt_w_f16(const signed char* __restrict__ w, _Float16* __restrict__ wb)
{
    const size_t i0 = ((size_t)blockIdx.x * 256 + threadIdx.x) * 16;
    int4 b = *(const int4*)(w + i0);
    const signed char* bb = (const signed char*)&b;
    v8h h0, h1;
#pragma unroll
    for (int j = 0; j < 8; ++j) {
        h0[j] = (_Float16)(int)bb[j];
        h1[j] = (_Float16)(int)bb[j + 8];
    }
    *(v8h*)(wb + i0)     = h0;
    *(v8h*)(wb + i0 + 8) = h1;
}

// ---------------------------------------------------------------------------
// Pass 2: pure-f16 GEMM. 128x128 block tile, 8 wave32s (2M x 4N wave grid),
// each wave 64x32 = 4x2 WMMA frags; BK=64 slab double-buffered in LDS.
// ---------------------------------------------------------------------------
__global__ __launch_bounds__(256)
void gemm_f16_wmma(const _Float16* __restrict__ xa,
                   const _Float16* __restrict__ wb,
                   const float* __restrict__ scales,
                   const float* __restrict__ bias,
                   float* __restrict__ out)
{
    __shared__ _Float16 lds_a[2][BM][LDS_LD];   // A: [row][k]
    __shared__ _Float16 lds_b[2][BN][LDS_LD];   // B: [col][k] (K-contiguous)

    const int tid   = threadIdx.x;
    const int lane  = tid & 31;
    const int wave  = tid >> 5;
    const int waveM = wave & 1;
    const int waveN = wave >> 1;
    const int half  = lane >> 4;
    const int l15   = lane & 15;

    const int blkN0 = blockIdx.x * BN;
    const int blkM0 = blockIdx.y * BM;

    // Staging: slab is 128 rows x 64 f16 = 16 KB per matrix.
    // Per thread: 4 x v8h chunks each for A and B (fi -> row = fi/8, cg = fi%8).
    int s_r[4], s_c[4];
#pragma unroll
    for (int i = 0; i < 4; ++i) {
        const int fi = tid + i * 256;
        s_r[i] = fi >> 3;
        s_c[i] = fi & 7;
    }
    v8h areg[4], brg[4];

    auto load_tile = [&](int kt) {
        const int k0 = kt * BK;
#pragma unroll
        for (int i = 0; i < 4; ++i) {
            areg[i] = *(const v8h*)(xa + (size_t)(blkM0 + s_r[i]) * DIN + k0 + s_c[i] * 8);
            brg[i]  = *(const v8h*)(wb + (size_t)(blkN0 + s_r[i]) * DIN + k0 + s_c[i] * 8);
        }
    };
    auto store_tile = [&](int buf) {
#pragma unroll
        for (int i = 0; i < 4; ++i) {
            *(v8h*)&lds_a[buf][s_r[i]][s_c[i] * 8] = areg[i];
            *(v8h*)&lds_b[buf][s_r[i]][s_c[i] * 8] = brg[i];
        }
    };

    v8f acc[4][2] = {};

    load_tile(0);
    store_tile(0);
    __syncthreads();

    for (int kt = 0; kt < NK; ++kt) {
        const int buf = kt & 1;

        if (kt + 1 < NK) load_tile(kt + 1);   // hide under 16 WMMAs
        if (kt + 2 < NK) {
            __builtin_prefetch(wb + (size_t)(blkN0 + s_r[0]) * DIN + (size_t)(kt + 2) * BK, 0, 1);
            __builtin_prefetch(xa + (size_t)(blkM0 + s_r[0]) * DIN + (size_t)(kt + 2) * BK, 0, 1);
        }

#pragma unroll
        for (int ks = 0; ks < 2; ++ks) {
            const int kb = ks * 32;

            FragU A[4];
#pragma unroll
            for (int mt = 0; mt < 4; ++mt) {
                const _Float16* p = &lds_a[buf][waveM * 64 + mt * 16 + l15][kb + half * 8];
                A[mt].h[0] = *(const v8h*)(p);        // K = kb + half*8 .. +7
                A[mt].h[1] = *(const v8h*)(p + 16);   // K = kb + 16 + half*8 .. +7
            }
            FragU Bf[2];
#pragma unroll
            for (int nt = 0; nt < 2; ++nt) {
                const _Float16* p = &lds_b[buf][waveN * 32 + nt * 16 + l15][kb + half * 8];
                Bf[nt].h[0] = *(const v8h*)(p);       // K = kb + half*8 .. +7
                Bf[nt].h[1] = *(const v8h*)(p + 16);  // K = kb + 16 + half*8 .. +7
            }

#pragma unroll
            for (int mt = 0; mt < 4; ++mt)
#pragma unroll
                for (int nt = 0; nt < 2; ++nt)
                    acc[mt][nt] = __builtin_amdgcn_wmma_f32_16x16x32_f16(
                        false, A[mt].v, false, Bf[nt].v,
                        (short)0, acc[mt][nt], false, false);
        }

        if (kt + 1 < NK) store_tile((kt + 1) & 1);
        __syncthreads();
    }

#pragma unroll
    for (int nt = 0; nt < 2; ++nt) {
        const int col = blkN0 + waveN * 32 + nt * 16 + l15;
        const float sc = scales[col];
        const float bv = bias[col];
#pragma unroll
        for (int mt = 0; mt < 4; ++mt) {
            const int row0 = blkM0 + waveM * 64 + mt * 16 + half * 8;
#pragma unroll
            for (int i = 0; i < 8; ++i)
                out[(size_t)(row0 + i) * DOUT + col] = acc[mt][nt][i] * sc + bv;
        }
    }
}

// ---------------------------------------------------------------------------
// Fallback: fused conversion-in-loop GEMM (no workspace requirement).
// ---------------------------------------------------------------------------
__global__ __launch_bounds__(256)
void int8_linear_wmma_fused(const float* __restrict__ x,
                            const signed char* __restrict__ w,
                            const float* __restrict__ scales,
                            const float* __restrict__ bias,
                            float* __restrict__ out)
{
    __shared__ _Float16 lds_a[2][BM][LDS_LD];
    __shared__ _Float16 lds_b[2][BN][LDS_LD];

    const int tid   = threadIdx.x;
    const int lane  = tid & 31;
    const int wave  = tid >> 5;
    const int waveM = wave & 1;
    const int waveN = wave >> 1;
    const int half  = lane >> 4;
    const int l15   = lane & 15;

    const int blkN0 = blockIdx.x * BN;
    const int blkM0 = blockIdx.y * BM;

    float4 areg[8];
    int4   breg[2];

    int a_r[8], a_c[8];
#pragma unroll
    for (int i = 0; i < 8; ++i) {
        const int fi = tid + i * 256;
        a_r[i] = fi >> 4;
        a_c[i] = fi & 15;
    }
    int b_n[2], b_ch[2];
#pragma unroll
    for (int i = 0; i < 2; ++i) {
        const int bi = tid + i * 256;
        b_n[i]  = bi >> 2;
        b_ch[i] = bi & 3;
    }

    auto load_tile = [&](int kt) {
        const int k0 = kt * BK;
#pragma unroll
        for (int i = 0; i < 8; ++i)
            areg[i] = *(const float4*)(x + (size_t)(blkM0 + a_r[i]) * DIN + k0 + a_c[i] * 4);
#pragma unroll
        for (int i = 0; i < 2; ++i)
            breg[i] = *(const int4*)(w + (size_t)(blkN0 + b_n[i]) * DIN + k0 + b_ch[i] * 16);
    };
    auto store_tile = [&](int buf) {
#pragma unroll
        for (int i = 0; i < 8; ++i) {
            v4h h;
            h[0] = (_Float16)areg[i].x;
            h[1] = (_Float16)areg[i].y;
            h[2] = (_Float16)areg[i].z;
            h[3] = (_Float16)areg[i].w;
            *(v4h*)&lds_a[buf][a_r[i]][a_c[i] * 4] = h;
        }
#pragma unroll
        for (int i = 0; i < 2; ++i) {
            const signed char* bb = (const signed char*)&breg[i];
            v8h h0, h1;
#pragma unroll
            for (int j = 0; j < 8; ++j) {
                h0[j] = (_Float16)(int)bb[j];
                h1[j] = (_Float16)(int)bb[j + 8];
            }
            *(v8h*)&lds_b[buf][b_n[i]][b_ch[i] * 16 + 0] = h0;
            *(v8h*)&lds_b[buf][b_n[i]][b_ch[i] * 16 + 8] = h1;
        }
    };

    v8f acc[4][2] = {};

    load_tile(0);
    store_tile(0);
    __syncthreads();

    for (int kt = 0; kt < NK; ++kt) {
        const int buf = kt & 1;
        if (kt + 1 < NK) load_tile(kt + 1);
        if (kt + 2 < NK)
            __builtin_prefetch(w + (size_t)(blkN0 + b_n[0]) * DIN + (size_t)(kt + 2) * BK, 0, 1);

#pragma unroll
        for (int ks = 0; ks < 2; ++ks) {
            const int kb = ks * 32;
            FragU A[4];
#pragma unroll
            for (int mt = 0; mt < 4; ++mt) {
                const _Float16* p = &lds_a[buf][waveM * 64 + mt * 16 + l15][kb + half * 8];
                A[mt].h[0] = *(const v8h*)(p);
                A[mt].h[1] = *(const v8h*)(p + 16);
            }
            FragU Bf[2];
#pragma unroll
            for (int nt = 0; nt < 2; ++nt) {
                const _Float16* p = &lds_b[buf][waveN * 32 + nt * 16 + l15][kb + half * 8];
                Bf[nt].h[0] = *(const v8h*)(p);
                Bf[nt].h[1] = *(const v8h*)(p + 16);
            }
#pragma unroll
            for (int mt = 0; mt < 4; ++mt)
#pragma unroll
                for (int nt = 0; nt < 2; ++nt)
                    acc[mt][nt] = __builtin_amdgcn_wmma_f32_16x16x32_f16(
                        false, A[mt].v, false, Bf[nt].v,
                        (short)0, acc[mt][nt], false, false);
        }

        if (kt + 1 < NK) store_tile((kt + 1) & 1);
        __syncthreads();
    }

#pragma unroll
    for (int nt = 0; nt < 2; ++nt) {
        const int col = blkN0 + waveN * 32 + nt * 16 + l15;
        const float sc = scales[col];
        const float bv = bias[col];
#pragma unroll
        for (int mt = 0; mt < 4; ++mt) {
            const int row0 = blkM0 + waveM * 64 + mt * 16 + half * 8;
#pragma unroll
            for (int i = 0; i < 8; ++i)
                out[(size_t)(row0 + i) * DOUT + col] = acc[mt][nt][i] * sc + bv;
        }
    }
}

extern "C" void kernel_launch(void* const* d_in, const int* in_sizes, int n_in,
                              void* d_out, int out_size, void* d_ws, size_t ws_size,
                              hipStream_t stream) {
    (void)in_sizes; (void)n_in; (void)out_size;
    const float*       x      = (const float*)d_in[0];
    const signed char* w      = (const signed char*)d_in[1];
    const float*       scales = (const float*)d_in[2];
    const float*       bias   = (const float*)d_in[3];
    float*             out    = (float*)d_out;

    const size_t xa_bytes = (size_t)MTOTAL * DIN * sizeof(_Float16);  // 64 MiB
    const size_t wb_bytes = (size_t)DOUT   * DIN * sizeof(_Float16);  // 32 MiB

    dim3 grid(DOUT / BN, MTOTAL / BM);   // (32, 64)
    dim3 block(256);                     // 8 wave32s

    if (ws_size >= xa_bytes + wb_bytes) {
        _Float16* xa = (_Float16*)d_ws;
        _Float16* wb = (_Float16*)((char*)d_ws + xa_bytes);
        // Pass 1: one-time precision conversion (bandwidth-bound, ~6 us total).
        cvt_x_f16<<<(MTOTAL * (DIN / 8)) / 256, 256, 0, stream>>>(x, xa);
        cvt_w_f16<<<(DOUT * (DIN / 16)) / 256, 256, 0, stream>>>(w, wb);
        // Pass 2: pure-f16 WMMA GEMM.
        gemm_f16_wmma<<<grid, block, 0, stream>>>(xa, wb, scales, bias, out);
    } else {
        int8_linear_wmma_fused<<<grid, block, 0, stream>>>(x, w, scales, bias, out);
    }
}
